// designed_discriminator_50182397886562
// MI455X (gfx1250) — matrix operand order repaired and checked by reading, not compiled, not run on hardware
//
#include <hip/hip_runtime.h>
#include <hip/hip_bf16.h>

// ---------------------------------------------------------------------------
// MI455X (gfx1250) windowed attention:
//   x(16,4096,512) -> qkv GEMM -> per-row-window attention (8 heads, hd=64)
//   -> proj GEMM + bias.
// All matmuls on v_wmma_f32_16x16x32_bf16 (wave32), f32 accumulate.
// GEMM B-operands staged per-workgroup into LDS with async global->LDS copies
// (GLOBAL_LOAD_ASYNC_TO_LDS_B128 + s_wait_asynccnt).
// ---------------------------------------------------------------------------

typedef __attribute__((ext_vector_type(16))) __bf16 v16bf;
typedef __attribute__((ext_vector_type(8)))  __bf16 v8bf;
typedef __attribute__((ext_vector_type(8)))  float  v8f;
typedef __attribute__((ext_vector_type(4)))  int    v4i;

#define WMMA_BF16(a, b, c) \
  __builtin_amdgcn_wmma_f32_16x16x32_bf16(false, (a), false, (b), (short)0, (c), false, false)

// Problem constants
#define TOKENS   65536      // B*N = 16*4096
#define CDIM     512
#define K3C      1536
#define NWIN     1024       // B*H = 16*64
#define NHEAD    8
#define HDIM     64
#define WHH      (NWIN * NHEAD)   // 8192 window-head pairs
#define QKVW_ELEMS  (K3C * CDIM)      // 786432
#define PROJW_ELEMS (CDIM * CDIM)     // 262144

#define BPAD 272            // 256-K half-slab row + 16 elem pad (544B stride)

#if defined(__gfx1250__) && \
    __has_builtin(__builtin_amdgcn_global_load_async_to_lds_b128) && \
    __has_builtin(__builtin_amdgcn_s_wait_asynccnt)
#define ASYNC_STAGE 1
#else
#define ASYNC_STAGE 0
#endif

static __device__ __forceinline__ void stage_b128(const __bf16* g, __bf16* l) {
#if ASYNC_STAGE
  __builtin_amdgcn_global_load_async_to_lds_b128(
      (__attribute__((address_space(1))) v4i*)(__bf16*)g,
      (__attribute__((address_space(3))) v4i*)l, 0, 0);
#else
  *(v8bf*)l = *(const v8bf*)g;
#endif
}

static __device__ __forceinline__ void stage_wait() {
#if ASYNC_STAGE
  __builtin_amdgcn_s_wait_asynccnt(0);
#endif
}

static __device__ __forceinline__ v16bf combine8(v8bf lo, v8bf hi) {
  v16bf r;
#pragma unroll
  for (int i = 0; i < 8; ++i) { r[i] = lo[i]; r[8 + i] = hi[i]; }
  return r;
}

// A-fragment (16x32 bf16): this lane's 16 contiguous K elements are p[0..15]
static __device__ __forceinline__ v16bf load_a_bf16(const __bf16* p) {
  v8bf lo = *(const v8bf*)p;
  v8bf hi = *(const v8bf*)(p + 8);
  return combine8(lo, hi);
}

// B-fragment: 16 contiguous bf16 via two 16B loads (16B-aligned LDS rows)
static __device__ __forceinline__ v16bf load_b16(const __bf16* p) {
  v8bf lo = *(const v8bf*)p;
  v8bf hi = *(const v8bf*)(p + 8);
  return combine8(lo, hi);
}

// A-fragment from fp32 memory with on-the-fly bf16 conversion
static __device__ __forceinline__ v16bf load_a_f32(const float* p) {
  float4 c0 = ((const float4*)p)[0];
  float4 c1 = ((const float4*)p)[1];
  float4 c2 = ((const float4*)(p + 8))[0];
  float4 c3 = ((const float4*)(p + 8))[1];
  v16bf r;
  r[0]  = (__bf16)c0.x; r[1]  = (__bf16)c0.y; r[2]  = (__bf16)c0.z; r[3]  = (__bf16)c0.w;
  r[4]  = (__bf16)c1.x; r[5]  = (__bf16)c1.y; r[6]  = (__bf16)c1.z; r[7]  = (__bf16)c1.w;
  r[8]  = (__bf16)c2.x; r[9]  = (__bf16)c2.y; r[10] = (__bf16)c2.z; r[11] = (__bf16)c2.w;
  r[12] = (__bf16)c3.x; r[13] = (__bf16)c3.y; r[14] = (__bf16)c3.z; r[15] = (__bf16)c3.w;
  return r;
}

// ---------------------------------------------------------------------------
// Kernel 1: convert weights fp32 -> bf16 (done once, reused 4096x per tile)
// ---------------------------------------------------------------------------
__global__ void __launch_bounds__(256) prep_weights(
    const float* __restrict__ qkvw, const float* __restrict__ projw,
    __bf16* __restrict__ qkvwb, __bf16* __restrict__ projwb) {
  int i = blockIdx.x * 256 + threadIdx.x;
  if (i < QKVW_ELEMS)  qkvwb[i]  = (__bf16)qkvw[i];
  if (i < PROJW_ELEMS) projwb[i] = (__bf16)projw[i];
}

// ---------------------------------------------------------------------------
// Kernel 2: QKV GEMM.  out[m, n] = sum_k x[m,k] * qkv_w[n,k]
//   grid = (24 n-strips of 64, 1024 windows), block = 128 (4 waves).
//   B slab (64 x 512) staged into LDS in two 64x256 halves via async copies;
//   all 4 waves consume the shared slab. A streamed fp32 -> bf16 from global.
//   Each 64-wide N strip is exactly one (q|k|v, head); q/k written [w][d],
//   v written transposed [d][w] so P@V B-frags are contiguous loads later.
// ---------------------------------------------------------------------------
__global__ void __launch_bounds__(128) qkv_kernel(
    const float* __restrict__ x, const __bf16* __restrict__ wb,
    __bf16* __restrict__ qb, __bf16* __restrict__ kb, __bf16* __restrict__ vtb) {
  __shared__ __bf16 bl[64][BPAD];
  const int tid = threadIdx.x;
  const int lane = tid & 31;
  const int wave = tid >> 5;             // 0..3 -> 16-row strip within window
  const int lm = lane & 15;
  const int lh = lane >> 4;              // K-chunk half select
  const int n64 = blockIdx.x;            // 0..23: 64-wide output channel strip
  const int win = blockIdx.y;            // 0..1023: window (64 tokens)

  const float* xrow = x + (size_t)(win * 64 + wave * 16 + lm) * CDIM;
  const __bf16* wtile = wb + (size_t)n64 * 64 * CDIM;

  v8f acc[4] = {};
  for (int half = 0; half < 2; ++half) {
    if (half) __syncthreads();           // prior slab fully consumed
    // stage 64 rows x 256 K: 2048 b128 chunks, 16 per thread
#pragma unroll
    for (int it = 0; it < 16; ++it) {
      int c = it * 128 + tid;
      int row = c >> 5;
      int col = (c & 31) * 8;
      stage_b128(wtile + (size_t)row * CDIM + half * 256 + col, &bl[row][col]);
    }
    stage_wait();
    __syncthreads();

    for (int kkl = 0; kkl < 8; ++kkl) {
      v16bf a = load_a_f32(xrow + (half * 8 + kkl) * 32 + lh * 16);
#pragma unroll
      for (int t = 0; t < 4; ++t) {
        v16bf b = load_b16(&bl[t * 16 + lm][kkl * 32 + lh * 16]);
        acc[t] = WMMA_BF16(a, b, acc[t]);
      }
    }
  }

  const int kind = n64 >> 3;             // 0=q 1=k 2=v
  const int head = n64 & 7;
  __bf16* base = (kind == 0) ? qb : (kind == 1) ? kb : vtb;
  base += ((size_t)win * NHEAD + head) * (64 * HDIM);

  if (kind < 2) {
    // q/k: [w][d] row-major; lane holds column d = t*16+lm, rows w vary with r
#pragma unroll
    for (int t = 0; t < 4; ++t)
#pragma unroll
      for (int r = 0; r < 8; ++r)
        base[(size_t)(wave * 16 + lh * 8 + r) * HDIM + t * 16 + lm] = (__bf16)acc[t][r];
  } else {
    // v^T: [d][w]; for a lane, w varies with r -> 8 contiguous bf16 = one 16B store
#pragma unroll
    for (int t = 0; t < 4; ++t) {
      v8bf pack;
#pragma unroll
      for (int r = 0; r < 8; ++r) pack[r] = (__bf16)acc[t][r];
      *(v8bf*)(base + (size_t)(t * 16 + lm) * 64 + wave * 16 + lh * 8) = pack;
    }
  }
}

// ---------------------------------------------------------------------------
// Kernel 3: attention. One wave per window-head (8192 total).
//   Per 16-row strip: S = q k^T (8 WMMA), register softmax, P staged through a
//   padded per-wave LDS slice (C-layout -> A-fragment relayout), O = P v (8 WMMA).
// ---------------------------------------------------------------------------
__global__ void __launch_bounds__(256) attn_kernel(
    const __bf16* __restrict__ qb, const __bf16* __restrict__ kbuf,
    const __bf16* __restrict__ vtb, __bf16* __restrict__ ob) {
  __shared__ __bf16 plds[8][16][72];     // per-wave 16x64 strip, +8 pad vs bank conflicts
  const int lane = threadIdx.x & 31;
  const int wave = threadIdx.x >> 5;     // 0..7
  const int lm = lane & 15;
  const int lh = lane >> 4;
  const int whh = blockIdx.x * 8 + wave; // window-head pair
  const int win = whh >> 3;
  const int head = whh & 7;

  const __bf16* q  = qb   + (size_t)whh * (64 * HDIM);
  const __bf16* k  = kbuf + (size_t)whh * (64 * HDIM);
  const __bf16* vt = vtb  + (size_t)whh * (64 * HDIM);
  __bf16* outp = ob + (size_t)win * 64 * CDIM + head * HDIM;
  const float scale = 0.125f;            // hd^-0.5 = 1/8

  for (int strip = 0; strip < 4; ++strip) {
    const int m0 = strip * 16;
    v8f s[4] = {};
#pragma unroll
    for (int kk = 0; kk < 2; ++kk) {
      v16bf a = load_a_bf16(q + (size_t)(m0 + lm) * HDIM + kk * 32 + lh * 16);
#pragma unroll
      for (int t = 0; t < 4; ++t) {
        v16bf b = *(const v16bf*)(k + (size_t)(t * 16 + lm) * HDIM + kk * 32 + lh * 16);
        s[t] = WMMA_BF16(a, b, s[t]);
      }
    }
    // Row softmax: row m = lh*8+r spans 4 tiles x 16 lanes; xor masks <16 stay in-half.
#pragma unroll
    for (int r = 0; r < 8; ++r) {
      float mx = -3.4e38f;
#pragma unroll
      for (int t = 0; t < 4; ++t) mx = fmaxf(mx, s[t][r] * scale);
#pragma unroll
      for (int off = 1; off < 16; off <<= 1) mx = fmaxf(mx, __shfl_xor(mx, off, 32));
      float sum = 0.f;
#pragma unroll
      for (int t = 0; t < 4; ++t) {
        float e = __expf(s[t][r] * scale - mx);
        s[t][r] = e;
        sum += e;
      }
#pragma unroll
      for (int off = 1; off < 16; off <<= 1) sum += __shfl_xor(sum, off, 32);
      float inv = 1.0f / sum;
#pragma unroll
      for (int t = 0; t < 4; ++t) s[t][r] *= inv;
    }
    // Relayout P: C-layout regs -> row-major LDS -> A-fragments
#pragma unroll
    for (int t = 0; t < 4; ++t)
#pragma unroll
      for (int r = 0; r < 8; ++r)
        plds[wave][lh * 8 + r][t * 16 + lm] = (__bf16)s[t][r];
    __syncthreads();

    v8f o[4] = {};
#pragma unroll
    for (int kk = 0; kk < 2; ++kk) {
      v16bf a = load_a_bf16(&plds[wave][lm][kk * 32 + lh * 16]);
#pragma unroll
      for (int t = 0; t < 4; ++t) {
        v16bf b = *(const v16bf*)(vt + (size_t)(t * 16 + lm) * 64 + kk * 32 + lh * 16);
        o[t] = WMMA_BF16(a, b, o[t]);
      }
    }
    __syncthreads();

    // out[token = win*64 + m0 + m, channel = head*64 + d]
#pragma unroll
    for (int t = 0; t < 4; ++t)
#pragma unroll
      for (int r = 0; r < 8; ++r)
        outp[(size_t)(m0 + lh * 8 + r) * CDIM + t * 16 + lm] = (__bf16)o[t][r];
  }
}

// ---------------------------------------------------------------------------
// Kernel 4: projection GEMM + bias, fp32 output. Same staged-B scheme.
//   grid = (8 n-strips, 1024 row blocks), block = 128 (4 waves, 16x64 each)
// ---------------------------------------------------------------------------
__global__ void __launch_bounds__(128) proj_kernel(
    const __bf16* __restrict__ ob, const __bf16* __restrict__ wb,
    const float* __restrict__ bias, float* __restrict__ out) {
  __shared__ __bf16 bl[64][BPAD];
  const int tid = threadIdx.x;
  const int lane = tid & 31;
  const int wave = tid >> 5;
  const int lm = lane & 15;
  const int lh = lane >> 4;
  const int n0 = blockIdx.x * 64;
  const int row0 = blockIdx.y * 64 + wave * 16;

  const __bf16* arow = ob + (size_t)(row0 + lm) * CDIM;
  const __bf16* wtile = wb + (size_t)n0 * CDIM;

  v8f acc[4] = {};
  for (int half = 0; half < 2; ++half) {
    if (half) __syncthreads();
#pragma unroll
    for (int it = 0; it < 16; ++it) {
      int c = it * 128 + tid;
      int row = c >> 5;
      int col = (c & 31) * 8;
      stage_b128(wtile + (size_t)row * CDIM + half * 256 + col, &bl[row][col]);
    }
    stage_wait();
    __syncthreads();

    for (int kkl = 0; kkl < 8; ++kkl) {
      v16bf a = load_a_bf16(arow + (half * 8 + kkl) * 32 + lh * 16);
#pragma unroll
      for (int t = 0; t < 4; ++t) {
        v16bf b = load_b16(&bl[t * 16 + lm][kkl * 32 + lh * 16]);
        acc[t] = WMMA_BF16(a, b, acc[t]);
      }
    }
  }
#pragma unroll
  for (int t = 0; t < 4; ++t) {
    float bn = bias[n0 + t * 16 + lm];
#pragma unroll
    for (int r = 0; r < 8; ++r)
      out[(size_t)(row0 + lh * 8 + r) * CDIM + n0 + t * 16 + lm] = acc[t][r] + bn;
  }
}

// ---------------------------------------------------------------------------
// Launch
// ---------------------------------------------------------------------------
extern "C" void kernel_launch(void* const* d_in, const int* in_sizes, int n_in,
                              void* d_out, int out_size, void* d_ws, size_t ws_size,
                              hipStream_t stream) {
  const float* x     = (const float*)d_in[0];
  const float* qkvw  = (const float*)d_in[1];
  const float* projw = (const float*)d_in[2];
  const float* projb = (const float*)d_in[3];
  float* out = (float*)d_out;

  char* ws = (char*)d_ws;
  size_t off = 0;
  auto carve = [&](size_t bytes) -> char* {
    char* p = ws + off;
    off = (off + bytes + 255) & ~(size_t)255;
    return p;
  };
  __bf16* qkvwb  = (__bf16*)carve((size_t)QKVW_ELEMS * 2);
  __bf16* projwb = (__bf16*)carve((size_t)PROJW_ELEMS * 2);
  __bf16* qb  = (__bf16*)carve((size_t)WHH * 64 * HDIM * 2);
  __bf16* kb  = (__bf16*)carve((size_t)WHH * 64 * HDIM * 2);
  __bf16* vtb = (__bf16*)carve((size_t)WHH * 64 * HDIM * 2);
  __bf16* ob  = (__bf16*)carve((size_t)TOKENS * CDIM * 2);

  prep_weights<<<(QKVW_ELEMS + 255) / 256, 256, 0, stream>>>(qkvw, projw, qkvwb, projwb);
  qkv_kernel<<<dim3(24, NWIN), 128, 0, stream>>>(x, qkvwb, qb, kb, vtb);
  attn_kernel<<<WHH / 8, 256, 0, stream>>>(qb, kb, vtb, ob);
  proj_kernel<<<dim3(8, 1024), 128, 0, stream>>>(ob, projwb, projb, out);
}
